// DeepSeekV3_1DecoderBlock_2001454760765
// MI455X (gfx1250) — matrix operand order repaired and checked by reading, not compiled, not run on hardware
//
#include <hip/hip_runtime.h>

// ---------------------------------------------------------------------------
// DeepSeek-V3 MLA decoder block for gfx1250 (MI455X, wave32, WMMA bf16)
// All heavy matmuls: v_wmma_f32_16x16x32_bf16, f32 accumulate.
// GEMM: 128x256 block tile, 32x128 wave tile, double-buffered LDS,
// unguarded global_load_b128 (weights padded to tile multiples), prefetch.
// ---------------------------------------------------------------------------

typedef __bf16 bf16;
typedef __attribute__((ext_vector_type(16))) __bf16 v16bf;
typedef __attribute__((ext_vector_type(8)))  __bf16 v8bf;
typedef __attribute__((ext_vector_type(8)))  float  v8f;

#define WMMA_BF16(A, B, C) \
  __builtin_amdgcn_wmma_f32_16x16x32_bf16(false, (A), false, (B), (short)0, (C), false, false)

static constexpr int S   = 2048;
static constexpr int D   = 2048;
static constexpr int H   = 16;
static constexpr int QL  = 1536;
static constexpr int DN  = 128;
static constexpr int DR  = 64;
static constexpr int DV  = 128;
static constexpr int KVL = 512;
static constexpr int F   = 8192;
static constexpr int HD    = DN + DR;      // 192
static constexpr int QOUT  = H * HD;       // 3072
static constexpr int KVOUT = H * (DN+DV);  // 4096
static constexpr int KVA   = KVL + DR;     // 576
static constexpr int KVA_P = 768;          // padded to 256-multiple for GEMM loads

// ---------------------------------------------------------------------------
// Weight cast + transpose: src[K][N] f32 -> dst[Nalloc][K] bf16, rows
// [N, Nalloc) are zero-filled so the GEMM can load without guards.
// ---------------------------------------------------------------------------
__global__ __launch_bounds__(256) void transpose_cast_bf16(
    const float* __restrict__ src, bf16* __restrict__ dst, int K, int N, int Nalloc)
{
  __shared__ float tile[32][33];
  int tx = threadIdx.x & 31, ty = threadIdx.x >> 5;   // 32 x 8
  int n = blockIdx.x * 32 + tx;
#pragma unroll
  for (int i = 0; i < 32; i += 8) {
    int k = blockIdx.y * 32 + ty + i;
    tile[ty + i][tx] = (k < K && n < N) ? src[(size_t)k * N + n] : 0.0f;
  }
  __syncthreads();
  int k2 = blockIdx.y * 32 + tx;
#pragma unroll
  for (int i = 0; i < 32; i += 8) {
    int n2 = blockIdx.x * 32 + ty + i;
    if (n2 < Nalloc && k2 < K) dst[(size_t)n2 * K + k2] = (bf16)tile[tx][ty + i];
  }
}

// ---------------------------------------------------------------------------
// RMSNorm (f32 in, bf16 out): one row per block
// ---------------------------------------------------------------------------
__global__ __launch_bounds__(256) void rmsnorm_bf16(
    const float* __restrict__ in, int in_stride, int L,
    const float* __restrict__ scale, bf16* __restrict__ out, int out_stride)
{
  const float* x = in + (size_t)blockIdx.x * in_stride;
  __shared__ float red[8];
  __shared__ float s_ms;
  float ss = 0.0f;
  for (int i = threadIdx.x; i < L; i += 256) { float v = x[i]; ss += v * v; }
#pragma unroll
  for (int o = 16; o > 0; o >>= 1) ss += __shfl_xor(ss, o, 32);
  if ((threadIdx.x & 31) == 0) red[threadIdx.x >> 5] = ss;
  __syncthreads();
  if (threadIdx.x == 0) {
    float t = 0.0f;
#pragma unroll
    for (int i = 0; i < 8; ++i) t += red[i];
    s_ms = t / (float)L;
  }
  __syncthreads();
  float r = rsqrtf(s_ms + 1e-6f);
  bf16* o = out + (size_t)blockIdx.x * out_stride;
  for (int i = threadIdx.x; i < L; i += 256) o[i] = (bf16)(x[i] * r * scale[i]);
}

// ---------------------------------------------------------------------------
// WMMA bf16 GEMM: C[M][N] = A[M][K] * Bt[N][K]^T, double-buffered LDS.
//   Requirements: M % 128 == 0, K % 32 == 0, Bt has >= ceil(N/256)*256 rows.
//   epilogues: +bias[col], +residual (f32 store);
//              gate != null: store bf16 silu(gate)*(acc+bias) to CB.
// Block tile 128x256x32, 8 waves (4 along M x 2 along N),
// wave tile 32x128 (2x8 WMMA accumulators); B fragment reused across both
// A fragments -> 16 WMMA per 20 ds_load_b128 per wave per K-step.
// ---------------------------------------------------------------------------
#define GBM 128
#define GBN 256
#define GBK 32
#define GLDS 40   // 32 + 8 halves padding (bank-conflict-free: 20 dwords)

__global__ __launch_bounds__(256) void gemm_bf16_wmma(
    const bf16* __restrict__ A, const bf16* __restrict__ Bt,
    float* __restrict__ C, bf16* __restrict__ CB,
    const float* __restrict__ bias, const float* __restrict__ residual,
    const float* __restrict__ gate, int M, int N, int K)
{
  __shared__ __align__(16) bf16 As[2][GBM * GLDS];   // 2 x 10 KB
  __shared__ __align__(16) bf16 Bs[2][GBN * GLDS];   // 2 x 20 KB
  int tid = threadIdx.x;
  int wave = tid >> 5, lane = tid & 31, ll = lane & 15, hi = lane >> 4;
  int wm = wave & 3, wn = wave >> 2;
  int m0 = blockIdx.y * GBM;
  int n0 = blockIdx.x * GBN;

  // per-thread staging coordinates: A rows r0,r0+64; B rows r0,+64,+128,+192
  int r0 = tid >> 2, c0 = (tid & 3) * 8;
  const bf16* pA = A  + (size_t)(m0 + r0) * K + c0;
  const bf16* pB = Bt + (size_t)(n0 + r0) * K + c0;
  const size_t rowK64 = (size_t)64 * K;
  int sOff = r0 * GLDS + c0;

  v8f acc[2][8] = {};

  // prologue: stage tile 0
  v8bf a0 = *(const v8bf*)pA;
  v8bf a1 = *(const v8bf*)(pA + rowK64);
  v8bf b0 = *(const v8bf*)pB;
  v8bf b1 = *(const v8bf*)(pB + rowK64);
  v8bf b2 = *(const v8bf*)(pB + 2 * rowK64);
  v8bf b3 = *(const v8bf*)(pB + 3 * rowK64);
  *(v8bf*)(&As[0][sOff])              = a0;
  *(v8bf*)(&As[0][sOff + 64 * GLDS])  = a1;
  *(v8bf*)(&Bs[0][sOff])              = b0;
  *(v8bf*)(&Bs[0][sOff + 64 * GLDS])  = b1;
  *(v8bf*)(&Bs[0][sOff + 128 * GLDS]) = b2;
  *(v8bf*)(&Bs[0][sOff + 192 * GLDS]) = b3;

  const int nk = K / GBK;
  int cur = 0;
  for (int kt = 0; kt < nk; ++kt) {
    __syncthreads();
    bool more = (kt + 1 < nk);
    if (more) {
      pA += GBK; pB += GBK;
      a0 = *(const v8bf*)pA;
      a1 = *(const v8bf*)(pA + rowK64);
      b0 = *(const v8bf*)pB;
      b1 = *(const v8bf*)(pB + rowK64);
      b2 = *(const v8bf*)(pB + 2 * rowK64);
      b3 = *(const v8bf*)(pB + 3 * rowK64);
      __builtin_prefetch(pA + GBK, 0, 1);          // global_prefetch_b8
      __builtin_prefetch(pB + GBK, 0, 1);
    }

    union Frag { v16bf v; v8bf h[2]; };
    Frag af[2];
#pragma unroll
    for (int i = 0; i < 2; ++i) {       // A fragment: halves at hi*8 and 16+hi*8
      const bf16* p = &As[cur][(wm * 32 + i * 16 + ll) * GLDS + hi * 8];
      af[i].h[0] = *(const v8bf*)p;
      af[i].h[1] = *(const v8bf*)(p + 16);
    }
#pragma unroll
    for (int j = 0; j < 8; ++j) {       // B fragment: halves at hi*8 and 16+hi*8
      Frag bfm;
      const bf16* p = &Bs[cur][(wn * 128 + j * 16 + ll) * GLDS + hi * 8];
      bfm.h[0] = *(const v8bf*)p;
      bfm.h[1] = *(const v8bf*)(p + 16);
      acc[0][j] = WMMA_BF16(af[0].v, bfm.v, acc[0][j]);
      acc[1][j] = WMMA_BF16(af[1].v, bfm.v, acc[1][j]);
    }

    if (more) {
      int nxt = cur ^ 1;
      *(v8bf*)(&As[nxt][sOff])              = a0;
      *(v8bf*)(&As[nxt][sOff + 64 * GLDS])  = a1;
      *(v8bf*)(&Bs[nxt][sOff])              = b0;
      *(v8bf*)(&Bs[nxt][sOff + 64 * GLDS])  = b1;
      *(v8bf*)(&Bs[nxt][sOff + 128 * GLDS]) = b2;
      *(v8bf*)(&Bs[nxt][sOff + 192 * GLDS]) = b3;
      cur = nxt;
    }
  }

  // epilogue (C layout: lane = column, vgpr r -> row r + hi*8)
#pragma unroll
  for (int i = 0; i < 2; ++i) {
#pragma unroll
    for (int j = 0; j < 8; ++j) {
      int col = n0 + wn * 128 + j * 16 + ll;
      if (col >= N) continue;
      float bv = bias ? bias[col] : 0.0f;
#pragma unroll
      for (int r = 0; r < 8; ++r) {
        int row = m0 + wm * 32 + i * 16 + hi * 8 + r;
        size_t idx = (size_t)row * N + col;
        float v = acc[i][j][r] + bv;
        if (gate) {
          float g = gate[idx];
          float sg = g / (1.0f + __expf(-g));   // silu
          CB[idx] = (bf16)(sg * v);
        } else {
          if (residual) v += residual[idx];
          C[idx] = v;
        }
      }
    }
  }
}

// ---------------------------------------------------------------------------
// Attention prep: RoPE + head-split + V transpose, bf16 outputs
//   qb[h][s][192] (pre-scaled by 1/sqrt(192)), kf[h][s][192], vT[h][d][s]
// ---------------------------------------------------------------------------
__global__ __launch_bounds__(256) void attn_prep(
    const float* __restrict__ q,    // [S][3072]
    const float* __restrict__ kv,   // [S][4096]
    const float* __restrict__ kva,  // [S][576]
    bf16* __restrict__ qb, bf16* __restrict__ kf, bf16* __restrict__ vT)
{
  int s = blockIdx.x;
  int t = threadIdx.x;
  __shared__ float cosv[64], sinv[64], krr[64];
  if (t < 64) {
    int j = t & 31;
    float freq = __powf(10000.0f, -(2.0f * (float)j) / 64.0f);
    float ang = (float)s * freq;
    cosv[t] = __cosf(ang);
    sinv[t] = __sinf(ang);
  }
  __syncthreads();
  if (t < 64) {
    float x1  = kva[(size_t)s * KVA + KVL + t];
    float rot = (t < 32) ? -kva[(size_t)s * KVA + KVL + t + 32]
                         :  kva[(size_t)s * KVA + KVL + t - 32];
    krr[t] = x1 * cosv[t] + rot * sinv[t];
  }
  __syncthreads();
  const float qscale = 0.07216878364870323f;  // 1/sqrt(192)
  for (int idx = t; idx < H * HD; idx += 256) {
    int h = idx / HD, d = idx % HD;
    float qv;
    if (d < DN) {
      qv = q[(size_t)s * QOUT + h * HD + d];
    } else {
      int p = d - DN;
      float x1  = q[(size_t)s * QOUT + h * HD + DN + p];
      float rot = (p < 32) ? -q[(size_t)s * QOUT + h * HD + DN + p + 32]
                           :  q[(size_t)s * QOUT + h * HD + DN + p - 32];
      qv = x1 * cosv[p] + rot * sinv[p];
    }
    qb[((size_t)h * S + s) * HD + d] = (bf16)(qv * qscale);
    float kvv = (d < DN) ? kv[(size_t)s * KVOUT + h * (DN+DV) + d] : krr[d - DN];
    kf[((size_t)h * S + s) * HD + d] = (bf16)kvv;
  }
  for (int idx = t; idx < H * DV; idx += 256) {
    int h = idx >> 7, d = idx & 127;
    vT[((size_t)h * DV + d) * S + s] =
        (bf16)kv[(size_t)s * KVOUT + h * (DN+DV) + DN + d];
  }
}

// ---------------------------------------------------------------------------
// Flash attention (causal): 128 q-rows / block, 16 rows / wave, WMMA bf16.
// Q fragments register-resident; K / V^T double-buffered in LDS.
// ---------------------------------------------------------------------------
__global__ __launch_bounds__(256) void mla_attention(
    const bf16* __restrict__ qb, const bf16* __restrict__ kf,
    const bf16* __restrict__ vT, bf16* __restrict__ out /* [S][H*DV] */)
{
  int h  = blockIdx.y;
  int q0 = blockIdx.x * 128;
  int tid = threadIdx.x;
  int wave = tid >> 5, lane = tid & 31, ll = lane & 15, hi = lane >> 4;

  __shared__ __align__(16) bf16 ks[2][32 * 200];   // K block, stride 200
  __shared__ __align__(16) bf16 vts[2][128 * 40];  // V^T block, stride 40
  __shared__ __align__(16) bf16 ps[8 * 16 * 40];   // P staging per wave

  const bf16* qh = qb + (size_t)h * S * HD;
  const bf16* kh = kf + (size_t)h * S * HD;
  const bf16* vh = vT + (size_t)h * DV * S;
  int qrow = q0 + wave * 16;

  union Frag { v16bf v; v8bf h2[2]; };

  // Q fragments for this wave's 16 rows: resident in registers (6 x 8 VGPR)
  Frag qfrag[6];
#pragma unroll
  for (int dc = 0; dc < 6; ++dc) {
    const bf16* qp = qh + (size_t)(qrow + ll) * HD + dc * 32 + hi * 8;
    qfrag[dc].h2[0] = *(const v8bf*)qp;
    qfrag[dc].h2[1] = *(const v8bf*)(qp + 16);
  }

  // staging coordinates
  int kR[3], kC[3];
#pragma unroll
  for (int i = 0; i < 3; ++i) { int idx = tid + i * 256; kR[i] = idx / 24; kC[i] = (idx % 24) * 8; }
  int vR = tid >> 2, vC = (tid & 3) * 8;   // rows vR and vR+64

  v8f o[8] = {};
  float mrow[8], lrow[8];
#pragma unroll
  for (int r = 0; r < 8; ++r) { mrow[r] = -1e30f; lrow[r] = 0.0f; }

  bf16* psw = &ps[wave * 640];

  // prologue: stage k-block 0
  v8bf kreg[3], vreg[2];
#pragma unroll
  for (int i = 0; i < 3; ++i)
    kreg[i] = *(const v8bf*)(kh + (size_t)kR[i] * HD + kC[i]);
  vreg[0] = *(const v8bf*)(vh + (size_t)vR * S + vC);
  vreg[1] = *(const v8bf*)(vh + (size_t)(vR + 64) * S + vC);
#pragma unroll
  for (int i = 0; i < 3; ++i) *(v8bf*)(&ks[0][kR[i] * 200 + kC[i]]) = kreg[i];
  *(v8bf*)(&vts[0][vR * 40 + vC]) = vreg[0];
  *(v8bf*)(&vts[0][(vR + 64) * 40 + vC]) = vreg[1];

  const int nkb = (q0 + 128) / 32;  // causal: k <= q0+127
  int cur = 0;
  for (int kb = 0; kb < nkb; ++kb) {
    int k0 = kb * 32;
    __syncthreads();
    bool more = (kb + 1 < nkb);
    if (more) {
      int k0n = k0 + 32;
#pragma unroll
      for (int i = 0; i < 3; ++i)
        kreg[i] = *(const v8bf*)(kh + (size_t)(k0n + kR[i]) * HD + kC[i]);
      vreg[0] = *(const v8bf*)(vh + (size_t)vR * S + k0n + vC);
      vreg[1] = *(const v8bf*)(vh + (size_t)(vR + 64) * S + k0n + vC);
    }

    // S = Q K^T over 6 d-chunks of 32
    v8f sacc[2] = {};
#pragma unroll
    for (int dc = 0; dc < 6; ++dc) {
#pragma unroll
      for (int t2 = 0; t2 < 2; ++t2) {
        Frag ub;
        const bf16* kp = &ks[cur][(t2 * 16 + ll) * 200 + dc * 32 + hi * 8];
        ub.h2[0] = *(const v8bf*)kp;
        ub.h2[1] = *(const v8bf*)(kp + 16);
        sacc[t2] = WMMA_BF16(qfrag[dc].v, ub.v, sacc[t2]);
      }
    }

    // causal mask + online softmax (C layout: lane = col, vgpr r -> row r+hi*8)
    float mnew[8], fsc[8];
#pragma unroll
    for (int r = 0; r < 8; ++r) {
      int grow = qrow + hi * 8 + r;
      float mx = -1e30f;
#pragma unroll
      for (int t2 = 0; t2 < 2; ++t2) {
        int col = k0 + t2 * 16 + ll;
        float sv = sacc[t2][r];
        if (col > grow) sv = -1e30f;
        sacc[t2][r] = sv;
        mx = fmaxf(mx, sv);
      }
#pragma unroll
      for (int off = 1; off < 16; off <<= 1) mx = fmaxf(mx, __shfl_xor(mx, off, 32));
      mnew[r] = fmaxf(mrow[r], mx);
      fsc[r]  = __expf(mrow[r] - mnew[r]);
      mrow[r] = mnew[r];
    }
#pragma unroll
    for (int r = 0; r < 8; ++r) {
      float rs = 0.0f;
#pragma unroll
      for (int t2 = 0; t2 < 2; ++t2) {
        float p = __expf(sacc[t2][r] - mnew[r]);
        rs += p;
        psw[(hi * 8 + r) * 40 + t2 * 16 + ll] = (bf16)p;
      }
#pragma unroll
      for (int off = 1; off < 16; off <<= 1) rs += __shfl_xor(rs, off, 32);
      lrow[r] = lrow[r] * fsc[r] + rs;
    }
    // rescale O
#pragma unroll
    for (int j = 0; j < 8; ++j)
#pragma unroll
      for (int r = 0; r < 8; ++r) o[j][r] *= fsc[r];

    // O += P @ V (P as A fragment from LDS, V^T rows as B fragments)
    {
      Frag ua;
      const bf16* pp = &psw[ll * 40 + hi * 8];
      ua.h2[0] = *(const v8bf*)pp;
      ua.h2[1] = *(const v8bf*)(pp + 16);
#pragma unroll
      for (int j = 0; j < 8; ++j) {
        Frag ub;
        const bf16* vp = &vts[cur][(j * 16 + ll) * 40 + hi * 8];
        ub.h2[0] = *(const v8bf*)vp;
        ub.h2[1] = *(const v8bf*)(vp + 16);
        o[j] = WMMA_BF16(ua.v, ub.v, o[j]);
      }
    }

    if (more) {
      int nxt = cur ^ 1;
#pragma unroll
      for (int i = 0; i < 3; ++i) *(v8bf*)(&ks[nxt][kR[i] * 200 + kC[i]]) = kreg[i];
      *(v8bf*)(&vts[nxt][vR * 40 + vC]) = vreg[0];
      *(v8bf*)(&vts[nxt][(vR + 64) * 40 + vC]) = vreg[1];
      cur = nxt;
    }
  }

  // normalize + store bf16 [s][h*128 + d]
#pragma unroll
  for (int j = 0; j < 8; ++j) {
    int col = h * DV + j * 16 + ll;
#pragma unroll
    for (int r = 0; r < 8; ++r) {
      int row = qrow + hi * 8 + r;
      out[(size_t)row * (H * DV) + col] = (bf16)(o[j][r] / lrow[r]);
    }
  }
}

// ---------------------------------------------------------------------------
// Host-side orchestration
// ---------------------------------------------------------------------------
extern "C" void kernel_launch(void* const* d_in, const int* in_sizes, int n_in,
                              void* d_out, int out_size, void* d_ws, size_t ws_size,
                              hipStream_t stream)
{
  (void)in_sizes; (void)n_in; (void)out_size; (void)ws_size;
  const float* x_in  = (const float*)d_in[0];
  // d_in[1] = attention_mask (causal, handled analytically)
  const float* pre_attn_scale = (const float*)d_in[2];
  const float* wqa  = (const float*)d_in[3];
  const float* q_norm_scale   = (const float*)d_in[4];
  const float* wqb  = (const float*)d_in[5];
  const float* wkva = (const float*)d_in[6];
  const float* kv_norm_scale  = (const float*)d_in[7];
  const float* wkvb = (const float*)d_in[8];
  const float* wo   = (const float*)d_in[9];
  const float* pre_ffn_scale  = (const float*)d_in[10];
  const float* wg   = (const float*)d_in[11];
  const float* bg   = (const float*)d_in[12];
  const float* wu   = (const float*)d_in[13];
  const float* bu   = (const float*)d_in[14];
  const float* wd   = (const float*)d_in[15];
  const float* bd   = (const float*)d_in[16];
  float* out = (float*)d_out;

  char* base = (char*)d_ws;
  size_t off = 0;
  auto alloc = [&](size_t bytes) -> void* {
    void* p = base + off;
    off += (bytes + 255) & ~(size_t)255;
    return p;
  };

  // persistent bf16 weights (transposed) + persistent activations
  bf16* wqa_t  = (bf16*)alloc((size_t)QL * D * 2);
  bf16* wqb_t  = (bf16*)alloc((size_t)QOUT * QL * 2);
  bf16* wkva_t = (bf16*)alloc((size_t)KVA_P * D * 2);   // padded rows zeroed
  bf16* wkvb_t = (bf16*)alloc((size_t)KVOUT * KVL * 2);
  bf16* wo_t   = (bf16*)alloc((size_t)D * (H * DV) * 2);
  bf16* wg_t   = (bf16*)alloc((size_t)F * D * 2);
  bf16* wu_t   = (bf16*)alloc((size_t)F * D * 2);
  bf16* wd_t   = (bf16*)alloc((size_t)D * F * 2);
  bf16* n_bf   = (bf16*)alloc((size_t)S * D * 2);
  float* x2    = (float*)alloc((size_t)S * D * 4);
  bf16* attn_bf= (bf16*)alloc((size_t)S * (H * DV) * 2);

  // attention-phase activations (aliased by FFN phase afterwards)
  size_t act0 = off;
  float* qpre  = (float*)alloc((size_t)S * QL * 4);
  bf16* qn_bf  = (bf16*)alloc((size_t)S * QL * 2);
  float* qf    = (float*)alloc((size_t)S * QOUT * 4);
  float* kva   = (float*)alloc((size_t)S * KVA * 4);
  bf16* ckv_bf = (bf16*)alloc((size_t)S * KVL * 2);
  float* kvf   = (float*)alloc((size_t)S * KVOUT * 4);
  bf16* qb     = (bf16*)alloc((size_t)H * S * HD * 2);
  bf16* kfb    = (bf16*)alloc((size_t)H * S * HD * 2);
  bf16* vT     = (bf16*)alloc((size_t)H * DV * S * 2);
  // FFN phase aliases:
  off = act0;
  bf16* n2_bf  = (bf16*)alloc((size_t)S * D * 2);
  float* gbuf  = (float*)alloc((size_t)S * F * 4);
  bf16* hbuf   = (bf16*)alloc((size_t)S * F * 2);

  dim3 blk(256);
  auto tgrid = [](int K, int Np) { return dim3((Np + 31) / 32, (K + 31) / 32); };
  // 1) weight casts (transpose to [N][K] bf16)
  transpose_cast_bf16<<<tgrid(D, QL),       blk, 0, stream>>>(wqa,  wqa_t,  D,    QL,    QL);
  transpose_cast_bf16<<<tgrid(QL, QOUT),    blk, 0, stream>>>(wqb,  wqb_t,  QL,   QOUT,  QOUT);
  transpose_cast_bf16<<<tgrid(D, KVA_P),    blk, 0, stream>>>(wkva, wkva_t, D,    KVA,   KVA_P);
  transpose_cast_bf16<<<tgrid(KVL, KVOUT),  blk, 0, stream>>>(wkvb, wkvb_t, KVL,  KVOUT, KVOUT);
  transpose_cast_bf16<<<tgrid(H*DV, D),     blk, 0, stream>>>(wo,   wo_t,   H*DV, D,     D);
  transpose_cast_bf16<<<tgrid(D, F),        blk, 0, stream>>>(wg,   wg_t,   D,    F,     F);
  transpose_cast_bf16<<<tgrid(D, F),        blk, 0, stream>>>(wu,   wu_t,   D,    F,     F);
  transpose_cast_bf16<<<tgrid(F, D),        blk, 0, stream>>>(wd,   wd_t,   F,    D,     D);

  auto ggrid = [](int M, int N) { return dim3((N + GBN - 1) / GBN, (M + GBM - 1) / GBM); };

  // 2) pre-attn RMSNorm
  rmsnorm_bf16<<<S, blk, 0, stream>>>(x_in, D, D, pre_attn_scale, n_bf, D);
  // 3) q_pre = n @ wqa
  gemm_bf16_wmma<<<ggrid(S, QL), blk, 0, stream>>>(n_bf, wqa_t, qpre, nullptr,
      nullptr, nullptr, nullptr, S, QL, D);
  // 4) q_norm
  rmsnorm_bf16<<<S, blk, 0, stream>>>(qpre, QL, QL, q_norm_scale, qn_bf, QL);
  // 5) q = qn @ wqb
  gemm_bf16_wmma<<<ggrid(S, QOUT), blk, 0, stream>>>(qn_bf, wqb_t, qf, nullptr,
      nullptr, nullptr, nullptr, S, QOUT, QL);
  // 6) kva = n @ wkva   (Bt padded to 768 rows; store guard keeps N=576)
  gemm_bf16_wmma<<<ggrid(S, KVA), blk, 0, stream>>>(n_bf, wkva_t, kva, nullptr,
      nullptr, nullptr, nullptr, S, KVA, D);
  // 7) kv_norm on first 512 cols
  rmsnorm_bf16<<<S, blk, 0, stream>>>(kva, KVA, KVL, kv_norm_scale, ckv_bf, KVL);
  // 8) kv = ckv @ wkvb
  gemm_bf16_wmma<<<ggrid(S, KVOUT), blk, 0, stream>>>(ckv_bf, wkvb_t, kvf, nullptr,
      nullptr, nullptr, nullptr, S, KVOUT, KVL);
  // 9) rope + split + V^T
  attn_prep<<<S, blk, 0, stream>>>(qf, kvf, kva, qb, kfb, vT);
  // 10) attention
  mla_attention<<<dim3(S / 128, H), blk, 0, stream>>>(qb, kfb, vT, attn_bf);
  // 11) x2 = x + attn @ wo
  gemm_bf16_wmma<<<ggrid(S, D), blk, 0, stream>>>(attn_bf, wo_t, x2, nullptr,
      nullptr, x_in, nullptr, S, D, H * DV);
  // 12) pre-ffn RMSNorm
  rmsnorm_bf16<<<S, blk, 0, stream>>>(x2, D, D, pre_ffn_scale, n2_bf, D);
  // 13) g = n2 @ wg + bg
  gemm_bf16_wmma<<<ggrid(S, F), blk, 0, stream>>>(n2_bf, wg_t, gbuf, nullptr,
      bg, nullptr, nullptr, S, F, D);
  // 14) h = silu(g) * (n2 @ wu + bu)   (fused into up-proj epilogue, bf16 out)
  gemm_bf16_wmma<<<ggrid(S, F), blk, 0, stream>>>(n2_bf, wu_t, nullptr, hbuf,
      bu, nullptr, gbuf, S, F, D);
  // 15) out = x2 + h @ wd + bd
  gemm_bf16_wmma<<<ggrid(S, D), blk, 0, stream>>>(hbuf, wd_t, out, nullptr,
      bd, x2, nullptr, S, D, F);
}
